// GCN_VN_15350213116756
// MI455X (gfx1250) — matrix operand order repaired and checked by reading, not compiled, not run on hardware
//
#include <hip/hip_runtime.h>
#include <hip/hip_bf16.h>
#include <math.h>

// ---------------------------------------------------------------------------
// GCN with virtual node + GATv2 on MI455X (gfx1250), wave32, WMMA bf16 path.
// All big GEMMs go through v_wmma_f32_16x16x32_bf16 (16x16 tiles, K=32/step).
// Packed-B weight panels are staged in LDS once per block (8x less L2 traffic);
// B fragments are register-blocked so the 8 WMMAs per k-chunk issue
// back-to-back behind a single s_wait_dscnt.
// ---------------------------------------------------------------------------

typedef __attribute__((ext_vector_type(16))) __bf16 v16bf;
typedef __attribute__((ext_vector_type(8)))  float  v8f;

#define HDIM   128
#define HEADS  4
#define DHEAD  32
#define GNUM   256
#define H2DIM  256

__device__ __forceinline__ float gelu_f(float x) {
    return 0.5f * x * (1.0f + erff(x * 0.70710678118654752440f));
}

__device__ __forceinline__ void atomicMaxF(float* addr, float v) {
    // monotonic int/uint trick; buffer initialized to -inf
    if (v >= 0.0f) atomicMax((int*)addr, __float_as_int(v));
    else           atomicMin((unsigned int*)addr, (unsigned int)__float_as_int(v));
}

// ---------------------------------------------------------------------------
// Weight packing: W[K,O] f32 row-major -> bf16 tiles in a contiguous-k B-matrix
// lane layout. Tile t = kc*(O/16)+ot holds 512 bf16; element (lane,j):
//   col = ot*16 + (lane&15);  k = kc*32 + (lane>>4)*16 + j
// ---------------------------------------------------------------------------
__global__ void pack_w_k(const float* __restrict__ W, __bf16* __restrict__ out,
                         int K, int O) {
    int i = blockIdx.x * blockDim.x + threadIdx.x;
    if (i >= K * O) return;
    int t    = i >> 9;          // tile
    int r    = i & 511;
    int lane = r >> 4;
    int j    = r & 15;
    int OT   = O >> 4;
    int ot   = t % OT;
    int kt   = t / OT;
    int col  = ot * 16 + (lane & 15);
    int k    = kt * 32 + (lane >> 4) * 16 + j;
    out[i] = (__bf16)W[(size_t)k * O + col];
}

// ---------------------------------------------------------------------------
// WMMA GEMM: Out[M,O] = act(A[M,K] @ W + bias) (+ Res). One wave = 16 rows x
// 128 cols (8 accumulator tiles). blockDim=256 (8 waves). grid.y covers O/128.
// B panel (up to K=256 x 128 cols = 64KB bf16) staged in LDS per block.
// ---------------------------------------------------------------------------
__global__ __launch_bounds__(256)
void gemm_bf16_k(const float* __restrict__ A, const __bf16* __restrict__ Bp,
                 const float* __restrict__ bias, const float* __restrict__ Res,
                 float* __restrict__ Out, int M, int K, int O, int act) {
    __shared__ __bf16 sB[8 * 8 * 512];          // [kc][ct][512] bf16, 64 KB max
    const int OT = O >> 4;
    const int KC = K >> 5;                      // <= 8
    const int colBase = blockIdx.y * 8;         // in 16-col tiles

    // cooperative, coalesced staging of this block's B panel into LDS
    {
        const uint4* src = (const uint4*)Bp;    // 16B units; 64 units per tile
        uint4* dst = (uint4*)sB;
        const int total = KC * 8 * 64;
        for (int idx = threadIdx.x; idx < total; idx += 256) {
            int tileL   = idx >> 6;             // kc*8 + ct
            int within  = idx & 63;
            int kc = tileL >> 3, ct = tileL & 7;
            dst[idx] = src[(size_t)(kc * OT + colBase + ct) * 64 + within];
        }
    }
    __syncthreads();

    const int lane = threadIdx.x & 31;
    const int wave = threadIdx.x >> 5;
    const int rowTiles = M >> 4;
    const int rt = blockIdx.x * 8 + wave;
    if (rt >= rowTiles) return;                 // after barrier: safe

    const int half = lane >> 4;
    const int l16  = lane & 15;

    v8f acc[8];
#pragma unroll
    for (int t = 0; t < 8; ++t)
#pragma unroll
        for (int v = 0; v < 8; ++v) acc[t][v] = 0.0f;

    const int row = rt * 16 + l16;
    const float* arow = A + (size_t)row * K;
    const v16bf* bt = (const v16bf*)sB;

    for (int kc = 0; kc < KC; ++kc) {
        const float* ap = arow + kc * 32;
        if (kc + 1 < KC) __builtin_prefetch(ap + 32, 0, 0);   // global_prefetch_b8
        // A fragment (global, f32 -> bf16 in-register)
        float4 f0 = *(const float4*)(ap + half * 16);
        float4 f1 = *(const float4*)(ap + half * 16 + 4);
        float4 f2 = *(const float4*)(ap + half * 16 + 8);
        float4 f3 = *(const float4*)(ap + half * 16 + 12);
        // B fragments: burst-load all 8 before any WMMA
        v16bf bfrag[8];
#pragma unroll
        for (int ct = 0; ct < 8; ++ct)
            bfrag[ct] = bt[(kc * 8 + ct) * 32 + lane];        // ds_load_b128 x2
        v16bf a;
        a[0] = (__bf16)f0.x;  a[1] = (__bf16)f0.y;  a[2]  = (__bf16)f0.z;  a[3]  = (__bf16)f0.w;
        a[4] = (__bf16)f1.x;  a[5] = (__bf16)f1.y;  a[6]  = (__bf16)f1.z;  a[7]  = (__bf16)f1.w;
        a[8] = (__bf16)f2.x;  a[9] = (__bf16)f2.y;  a[10] = (__bf16)f2.z;  a[11] = (__bf16)f2.w;
        a[12] = (__bf16)f3.x; a[13] = (__bf16)f3.y; a[14] = (__bf16)f3.z;  a[15] = (__bf16)f3.w;
#pragma unroll
        for (int ct = 0; ct < 8; ++ct)
            acc[ct] = __builtin_amdgcn_wmma_f32_16x16x32_bf16(
                false, a, false, bfrag[ct], (short)0, acc[ct], false, false);
    }

#pragma unroll
    for (int ct = 0; ct < 8; ++ct) {
        const int col = (colBase + ct) * 16 + l16;
        const float bb = bias ? bias[col] : 0.0f;
#pragma unroll
        for (int v = 0; v < 8; ++v) {
            int r = rt * 16 + v + half * 8;
            float val = acc[ct][v] + bb;
            if (act) val = gelu_f(val);
            if (Res) val += Res[(size_t)r * O + col];
            Out[(size_t)r * O + col] = val;
        }
    }
}

// --------------------------- elementwise helpers ---------------------------
__global__ void fill_k(float* p, float v, int n) {
    int i = blockIdx.x * blockDim.x + threadIdx.x;
    if (i < n) p[i] = v;
}
__global__ void bias_fill_k(float* dst, const float* __restrict__ bias, int n) {
    int i = blockIdx.x * blockDim.x + threadIdx.x;
    if (i < n) dst[i] = bias[i & (HDIM - 1)];
}
__global__ void count_k(const int* __restrict__ batch, float* cnt, int N) {
    int i = blockIdx.x * blockDim.x + threadIdx.x;
    if (i < N) atomicAdd(cnt + batch[i], 1.0f);
}
__global__ void pool_sum_k(const float* __restrict__ h, const int* __restrict__ batch,
                           float* pool, int N) {
    int i = blockIdx.x * blockDim.x + threadIdx.x;
    if (i >= N * HDIM) return;
    int n = i >> 7, c = i & 127;
    atomicAdd(pool + (size_t)batch[n] * HDIM + c, h[i]);
}
__global__ void pool_fin_k(float* pool, const float* __restrict__ cnt,
                           const float* __restrict__ emb) {
    int i = blockIdx.x * blockDim.x + threadIdx.x;
    if (i >= GNUM * HDIM) return;
    int g = i >> 7;
    float v = pool[i] / fmaxf(cnt[g], 1.0f);
    if (emb) v += emb[i & 127];
    pool[i] = v;
}

// --------------------------- batch-norm (biased var) -----------------------
__global__ void bn_stats_k(const float* __restrict__ X, int Nr, int C,
                           float* s, float* q) {
    int c = threadIdx.x;                 // blockDim.x == C
    float ls = 0.0f, lq = 0.0f;
    for (int r = blockIdx.x; r < Nr; r += gridDim.x) {
        float v = X[(size_t)r * C + c];
        ls += v; lq += v * v;
    }
    atomicAdd(s + c, ls);
    atomicAdd(q + c, lq);
}
__global__ void bn_apply_k(float* X, int Nr, int C, const float* __restrict__ s,
                           const float* __restrict__ q, const float* __restrict__ g,
                           const float* __restrict__ b, int act) {
    int i = blockIdx.x * blockDim.x + threadIdx.x;
    if (i >= Nr * C) return;
    int c = i % C;
    float inv = 1.0f / (float)Nr;
    float mu  = s[c] * inv;
    float var = q[c] * inv - mu * mu;
    float y = g[c] * (X[i] - mu) * rsqrtf(var + 1e-5f) + b[c];
    if (act) y = gelu_f(y);
    X[i] = y;
}

// --------------------------- GATv2 edge kernels ----------------------------
__global__ void edge_logits_k(const int* __restrict__ ei, int E, int Etot,
                              const float* __restrict__ xl, const float* __restrict__ xr,
                              const float* __restrict__ att,
                              float* __restrict__ logits, float* maxb) {
    int i = blockIdx.x * blockDim.x + threadIdx.x;
    if (i >= Etot * HEADS) return;
    int e = i >> 2, hh = i & 3;
    int s, d;
    if (e < E) { s = ei[e]; d = ei[E + e]; } else { s = e - E; d = s; }
    const float* pl = xl + (size_t)s * HDIM + hh * DHEAD;
    const float* pr = xr + (size_t)d * HDIM + hh * DHEAD;
    const float* pa = att + hh * DHEAD;
    float acc = 0.0f;
#pragma unroll
    for (int t = 0; t < DHEAD; ++t) {
        float v = pl[t] + pr[t];
        v = v > 0.0f ? v : 0.2f * v;          // LeakyReLU(0.2)
        acc += pa[t] * v;
    }
    logits[i] = acc;
    atomicMaxF(maxb + (size_t)d * HEADS + hh, acc);
}
__global__ void edge_exp_k(const int* __restrict__ ei, int E, int Etot,
                           const float* __restrict__ logits, const float* __restrict__ maxb,
                           float* __restrict__ aexp, float* denom) {
    int i = blockIdx.x * blockDim.x + threadIdx.x;
    if (i >= Etot * HEADS) return;
    int e = i >> 2, hh = i & 3;
    int d = (e < E) ? ei[E + e] : (e - E);
    float a = expf(logits[i] - maxb[(size_t)d * HEADS + hh]);
    aexp[i] = a;
    atomicAdd(denom + (size_t)d * HEADS + hh, a);
}
__global__ void edge_scatter_k(const int* __restrict__ ei, int E, int Etot,
                               const float* __restrict__ aexp, const float* __restrict__ denom,
                               const float* __restrict__ xl, float* conv) {
    int i = blockIdx.x * blockDim.x + threadIdx.x;
    if (i >= Etot * HEADS) return;
    int e = i >> 2, hh = i & 3;
    int s, d;
    if (e < E) { s = ei[e]; d = ei[E + e]; } else { s = e - E; d = s; }
    float alpha = aexp[i] / fmaxf(denom[(size_t)d * HEADS + hh], 1e-16f);
    const float* pl = xl + (size_t)s * HDIM + hh * DHEAD;
    float* po = conv + (size_t)d * HDIM + hh * DHEAD;
#pragma unroll
    for (int t = 0; t < DHEAD; ++t) atomicAdd(po + t, alpha * pl[t]);
}

// pre[:, :128] = conv + h ; pre[:, 128:] = vn[batch]
__global__ void build_pre_k(const float* __restrict__ conv, const float* __restrict__ h,
                            const float* __restrict__ vn, const int* __restrict__ batch,
                            float* pre, int N) {
    int i = blockIdx.x * blockDim.x + threadIdx.x;
    if (i >= N * H2DIM) return;
    int n = i >> 8, c = i & 255;
    if (c < HDIM) pre[i] = conv[(size_t)n * HDIM + c] + h[(size_t)n * HDIM + c];
    else          pre[i] = vn[(size_t)batch[n] * HDIM + (c - HDIM)];
}

// ---------------------------------------------------------------------------
extern "C" void kernel_launch(void* const* d_in, const int* in_sizes, int n_in,
                              void* d_out, int out_size, void* d_ws, size_t ws_size,
                              hipStream_t stream) {
    const int N    = in_sizes[2];
    const int E    = in_sizes[1] / 2;
    const int Etot = E + N;

    auto F = [&](int i) { return (const float*)d_in[i]; };
    const float* x       = F(0);
    const int*   ei      = (const int*)d_in[1];
    const int*   batch   = (const int*)d_in[2];
    // params flattened in setup_inputs() dict insertion order:
    const float* vn_emb  = F(3);
    // preprocess: 4 l1.W, 5 l1.b, 6 l2.W, 7 l2.b
    // update_vn[i] base 8+i*8: l1.W,l1.b,bn1.g,bn1.b,l2.W,l2.b,bn2.g,bn2.b
    // convs[i]     base 24+i*6: Wl,bl,Wr,br,att,bias
    // ffnns[i]     base 36+i*4: l1.W,l1.b,l2.W,l2.b
    // bns[i]       base 44+i*2: g,b
    // postprocess: 48 l1.W, 49 l1.b, 50 l2.W, 51 l2.b

    // -------- workspace carving --------
    char* base = (char*)d_ws;
    size_t cur = 0;
    auto carve = [&](size_t bytes) -> void* {
        cur = (cur + 255) & ~(size_t)255;
        void* p = base + cur;
        cur += bytes;
        return p;
    };
    float* h    = (float*)carve((size_t)N * HDIM * 4);
    float* xl   = (float*)carve((size_t)N * HDIM * 4);
    float* xr   = (float*)carve((size_t)N * HDIM * 4);
    float* conv = (float*)carve((size_t)N * HDIM * 4);   // also reused as GEMM temp
    float* pre  = (float*)carve((size_t)N * H2DIM * 4);
    float* logits = (float*)carve((size_t)Etot * HEADS * 4);
    float* aexp   = (float*)carve((size_t)Etot * HEADS * 4);
    float* maxb   = (float*)carve((size_t)N * HEADS * 4);
    float* denom  = (float*)carve((size_t)N * HEADS * 4);
    float* cnt    = (float*)carve(GNUM * 4);
    float* pool   = (float*)carve(GNUM * HDIM * 4);
    float* vn1    = (float*)carve(GNUM * H2DIM * 4);
    float* vn2    = (float*)carve(GNUM * HDIM * 4);
    float* post1  = (float*)carve(GNUM * HDIM * 4);
    float* ssum   = (float*)carve(H2DIM * 4);
    float* ssq    = (float*)carve(H2DIM * 4);

    auto carveB = [&](int K, int O) { return (__bf16*)carve((size_t)K * O * 2); };
    __bf16* pk_pre1 = carveB(128, 128);
    __bf16* pk_pre2 = carveB(128, 128);
    __bf16* pk_Wl[2]  = { carveB(128, 128), carveB(128, 128) };
    __bf16* pk_Wr[2]  = { carveB(128, 128), carveB(128, 128) };
    __bf16* pk_uv1[2] = { carveB(128, 256), carveB(128, 256) };
    __bf16* pk_uv2[2] = { carveB(256, 128), carveB(256, 128) };
    __bf16* pk_f1[2]  = { carveB(256, 128), carveB(256, 128) };
    __bf16* pk_f2[2]  = { carveB(128, 128), carveB(128, 128) };
    __bf16* pk_po1 = carveB(128, 128);
    __bf16* pk_po2 = carveB(128, 128);

    auto grid1 = [](int n) { return dim3((n + 255) / 256); };
    auto pack = [&](const float* W, __bf16* dst, int K, int O) {
        pack_w_k<<<grid1(K * O), 256, 0, stream>>>(W, dst, K, O);
    };
    auto gemm = [&](const float* A, const __bf16* Bp, const float* bias,
                    const float* Res, float* Out, int M, int K, int O, int act) {
        dim3 g((M / 16 + 7) / 8, O / 128, 1);
        gemm_bf16_k<<<g, 256, 0, stream>>>(A, Bp, bias, Res, Out, M, K, O, act);
    };
    auto zero = [&](float* p, int n) { fill_k<<<grid1(n), 256, 0, stream>>>(p, 0.0f, n); };
    auto bn = [&](float* X, int Nr, int C, const float* g, const float* b, int act) {
        zero(ssum, C); zero(ssq, C);
        int nb = Nr < 512 ? Nr : 512;
        bn_stats_k<<<nb, C, 0, stream>>>(X, Nr, C, ssum, ssq);
        bn_apply_k<<<grid1(Nr * C), 256, 0, stream>>>(X, Nr, C, ssum, ssq, g, b, act);
    };

    // -------- pack all GEMM weights (bf16, B-matrix lane layout) --------
    pack(F(4), pk_pre1, 128, 128);  pack(F(6), pk_pre2, 128, 128);
    for (int i = 0; i < 2; ++i) {
        pack(F(24 + i * 6),     pk_Wl[i], 128, 128);
        pack(F(24 + i * 6 + 2), pk_Wr[i], 128, 128);
        pack(F(8 + i * 8),      pk_uv1[i], 128, 256);
        pack(F(8 + i * 8 + 4),  pk_uv2[i], 256, 128);
        pack(F(36 + i * 4),     pk_f1[i], 256, 128);
        pack(F(36 + i * 4 + 2), pk_f2[i], 128, 128);
    }
    pack(F(48), pk_po1, 128, 128);  pack(F(50), pk_po2, 128, 128);

    // -------- preprocess FFNN: x -> h --------
    gemm(x, pk_pre1, F(5), nullptr, conv, N, 128, 128, 1);   // GELU
    gemm(conv, pk_pre2, F(7), nullptr, h, N, 128, 128, 0);

    // -------- per-graph node counts (constant across hops) --------
    zero(cnt, GNUM);
    count_k<<<grid1(N), 256, 0, stream>>>(batch, cnt, N);

    for (int i = 0; i < 2; ++i) {
        // ---- virtual node: vn = ffnn_vn(vn_emb + mean_pool(h)) ----
        zero(pool, GNUM * HDIM);
        pool_sum_k<<<grid1(N * HDIM), 256, 0, stream>>>(h, batch, pool, N);
        pool_fin_k<<<grid1(GNUM * HDIM), 256, 0, stream>>>(pool, cnt, vn_emb);
        gemm(pool, pk_uv1[i], F(8 + i * 8 + 1), nullptr, vn1, GNUM, 128, 256, 0);
        bn(vn1, GNUM, 256, F(8 + i * 8 + 2), F(8 + i * 8 + 3), 1);   // BN + GELU
        gemm(vn1, pk_uv2[i], F(8 + i * 8 + 5), nullptr, vn2, GNUM, 256, 128, 0);
        bn(vn2, GNUM, 128, F(8 + i * 8 + 6), F(8 + i * 8 + 7), 1);   // BN + GELU

        // ---- GATv2 conv ----
        const int cb = 24 + i * 6;
        gemm(h, pk_Wl[i], F(cb + 1), nullptr, xl, N, 128, 128, 0);
        gemm(h, pk_Wr[i], F(cb + 3), nullptr, xr, N, 128, 128, 0);
        bias_fill_k<<<grid1(N * HDIM), 256, 0, stream>>>(conv, F(cb + 5), N * HDIM);
        fill_k<<<grid1(N * HEADS), 256, 0, stream>>>(maxb, -INFINITY, N * HEADS);
        zero(denom, N * HEADS);
        edge_logits_k<<<grid1(Etot * HEADS), 256, 0, stream>>>(ei, E, Etot, xl, xr,
                                                               F(cb + 4), logits, maxb);
        edge_exp_k<<<grid1(Etot * HEADS), 256, 0, stream>>>(ei, E, Etot, logits, maxb,
                                                            aexp, denom);
        edge_scatter_k<<<grid1(Etot * HEADS), 256, 0, stream>>>(ei, E, Etot, aexp, denom,
                                                                xl, conv);

        // ---- pre = BN(concat(conv+h, vn[batch])) ; h = FFNN(pre) + h ----
        build_pre_k<<<grid1(N * H2DIM), 256, 0, stream>>>(conv, h, vn2, batch, pre, N);
        bn(pre, N, 256, F(44 + i * 2), F(44 + i * 2 + 1), 0);
        gemm(pre, pk_f1[i], F(36 + i * 4 + 1), nullptr, conv, N, 256, 128, 1);  // GELU
        gemm(conv, pk_f2[i], F(36 + i * 4 + 3), h, h, N, 128, 128, 0);          // +residual
    }

    // -------- postprocess: FFNN(mean_pool(h)) -> d_out [G, OUT] --------
    zero(pool, GNUM * HDIM);
    pool_sum_k<<<grid1(N * HDIM), 256, 0, stream>>>(h, batch, pool, N);
    pool_fin_k<<<grid1(GNUM * HDIM), 256, 0, stream>>>(pool, cnt, nullptr);
    gemm(pool, pk_po1, F(49), nullptr, post1, GNUM, 128, 128, 1);
    gemm(post1, pk_po2, F(51), nullptr, (float*)d_out, GNUM, 128, 128, 0);
}
